// GNNModelPyG_12575664243275
// MI455X (gfx1250) — matrix operand order repaired and checked by reading, not compiled, not run on hardware
//
#include <hip/hip_runtime.h>
#include <hip/hip_bf16.h>

// ---------------------------------------------------------------------------
// GNN (NNConv x3 + pool + MLP head) for gfx1250 (MI455X), wave32 + WMMA.
//
// msg[e,o] = sum_{i,k} xs[e,i]*h[e,k]*W2[(i*48+o),k] cast as one GEMM
//   P[E x 6208] @ B[6208 x 48]
// P built on the fly (P[e,i*128+k] = xs[e,i]*h[e,k]; last 64 K rows carry the
// b2 bias term, zero padded); B pre-packed per layer into WMMA B-fragment
// layout. Each wave owns a 32-edge M-tile: two A fragments per K-chunk share
// each B fragment (6 WMMAs / 6 b128 loads) and the second A's v_pk_mul_f16s
// fill the WMMA->VALU WAR hazard slots. D scatter-adds into aggr[dst[e]].
// __launch_bounds__(128,1) lifts the VGPR cap so the h-row register cache
// stays in registers instead of spilling to scratch.
// ---------------------------------------------------------------------------

typedef __attribute__((ext_vector_type(16))) _Float16 v16h;
typedef __attribute__((ext_vector_type(8)))  _Float16 v8h;
typedef __attribute__((ext_vector_type(8)))  float    v8f;

#define N_NODES 8192
#define N_EDGES 65536
#define N_GRAPHS 256
#define F_EDGE 16
#define HC 48            // H_CONV == F_NODE
#define HE 128           // H_EDGE
#define H_LIN 256
#define KTOT 6144        // HC * HE
#define KTILES 192       // KTOT / 32
#define KTILES_ALL 194   // + 2 bias K-chunks (48 real + 16 pad)
#define BPACK_ELEMS (KTILES_ALL * 3 * 32 * 16)
#define TILE_E 32        // edges per wave (M-tile)
#define WPB 4            // waves per block (128 threads)

// ---------------- stage 1: h = LeakyReLU(edge_attr @ W1^T + b1) -> f16 ------
__global__ void edge_mlp1_kernel(const float* __restrict__ ea,
                                 const float* __restrict__ W1,
                                 const float* __restrict__ b1,
                                 _Float16* __restrict__ h) {
  __shared__ float w[HE * F_EDGE];
  __shared__ float bb[HE];
  for (int u = threadIdx.x; u < HE * F_EDGE; u += 256) w[u] = W1[u];
  if (threadIdx.x < HE) bb[threadIdx.x] = b1[threadIdx.x];
  __syncthreads();
  int gid = blockIdx.x * 256 + threadIdx.x;   // = e*128 + k, exact cover
  int e = gid >> 7, k = gid & 127;
  const float* er = ea + (size_t)e * F_EDGE;
  float acc = bb[k];
#pragma unroll
  for (int j = 0; j < F_EDGE; ++j) acc += er[j] * w[k * F_EDGE + j];
  acc = acc > 0.f ? acc : 0.01f * acc;        // LeakyReLU(0.01)
  h[gid] = (_Float16)acc;
}

// ---------------- pack B[6208 x 48] into per-fragment WMMA layout -----------
// Fragment (t,nt): per lane 16 f16 at Bpack[((t*3+nt)*32+lane)*16 + j].
// B layout (CDNA5 ISA sparse-B table): column = lane&15,
// lane half h holds K=t*32+8h+j (j<8), K=t*32+16+8h+(j-8) (j>=8).
__global__ void bpack_kernel(const float* __restrict__ W2,
                             const float* __restrict__ b2,
                             _Float16* __restrict__ Bpack) {
  int idx = blockIdx.x * 256 + threadIdx.x;
  if (idx >= BPACK_ELEMS) return;
  int j    = idx & 15;
  int lane = (idx >> 4) & 31;
  int tnt  = idx >> 9;                 // t*3 + nt
  int t = tnt / 3, nt = tnt - t * 3;
  int n  = nt * 16 + (lane & 15);
  int k  = ((j < 8) ? j : (j + 8)) + ((lane >> 4) * 8);
  int Kg = t * 32 + k;
  float v = 0.f;
  if (Kg < KTOT) {                     // W2''[(i,k), n] = W2[(i*48+n)*128 + k]
    int i = Kg >> 7, kk = Kg & 127;
    v = W2[(size_t)(i * HC + n) * HE + kk];
  } else {                             // bias rows: B[6144+i, n] = b2[i*48+n]
    int i = Kg - KTOT;
    if (i < HC) v = b2[i * HC + n];
  }
  Bpack[idx] = (_Float16)v;
}

// ---------------- main fused WMMA kernel: P @ B -> atomic scatter -----------
__global__ void __launch_bounds__(WPB * 32, 1)
edge_gemm_kernel(const float* __restrict__ xcur,
                 const _Float16* __restrict__ hbuf,
                 const _Float16* __restrict__ Bpack,
                 const int* __restrict__ src,
                 const int* __restrict__ dst,
                 float* __restrict__ aggr) {
  __shared__ alignas(32) _Float16 hTile[WPB][TILE_E * HE];   // 32 KB
  __shared__ alignas(32) _Float16 xsTile[WPB][TILE_E * HC];  // 12 KB
  __shared__ int dstTile[WPB][TILE_E];

  const int wave = threadIdx.x >> 5;
  const int lane = threadIdx.x & 31;
  const int tile = blockIdx.x * WPB + wave;            // 2048 tiles of 32 edges
  const int row  = lane & 15;                          // M-row (low half of tile)
  const int half = lane >> 4;                          // lane half (K split)

  // Stage h tile: 32 consecutive edges -> 8KB contiguous slab of hbuf.
  {
    const uint4* hs = (const uint4*)(hbuf + (size_t)tile * (TILE_E * HE));
    uint4* hd = (uint4*)hTile[wave];
    for (int u = lane; u < 512; u += 32) hd[u] = hs[u];
  }
  // Stage dst indices and gathered source features (f32 -> f16), 1 row/lane.
  dstTile[wave][lane] = dst[tile * TILE_E + lane];
  {
    const float* xr = xcur + (size_t)src[tile * TILE_E + lane] * HC;
    for (int j = 0; j < HC; ++j)
      xsTile[wave][lane * HC + j] = (_Float16)xr[j];
  }
  __syncthreads();   // uniform; EXEC all-ones below (WMMA requirement)

  v8f accL0 = {}, accL1 = {}, accL2 = {};
  v8f accH0 = {}, accH1 = {}, accH2 = {};
  const _Float16* hrowL = &hTile[wave][row * HE];
  const _Float16* hrowH = &hTile[wave][(row + 16) * HE];

  // Main K loop: i = node-feature index (xs scalar), 4 K-chunks of 32 per i.
  for (int i = 0; i < HC; ++i) {
    _Float16 xsL = xsTile[wave][row * HC + i];
    _Float16 xsH = xsTile[wave][(row + 16) * HC + i];
#pragma unroll
    for (int c = 0; c < 4; ++c) {
      const int kbase = c * 32;
      // A fragments (16x32 f16): lane-half layout K = j + (j>=8?8:0) + 8*half
      const _Float16* hpL = hrowL + kbase + half * 8;
      const _Float16* hpH = hrowH + kbase + half * 8;
      v8h l0 = *(const v8h*)(hpL);
      v8h l1 = *(const v8h*)(hpL + 16);
      v8h u0 = *(const v8h*)(hpH);
      v8h u1 = *(const v8h*)(hpH + 16);
      v16h aL, aH;
#pragma unroll
      for (int j = 0; j < 8; ++j) {
        aL[j] = xsL * l0[j]; aL[j + 8] = xsL * l1[j];
        aH[j] = xsH * u0[j]; aH[j + 8] = xsH * u1[j];
      }

      const int t = i * 4 + c;
      const _Float16* bp = Bpack + (size_t)t * 1536 + (unsigned)lane * 16;
      __builtin_prefetch(bp + 1536, 0, 1);             // next K-chunk of B
      v16h b0 = *(const v16h*)(bp);
      v16h b1 = *(const v16h*)(bp + 512);
      v16h b2f = *(const v16h*)(bp + 1024);
      accL0 = __builtin_amdgcn_wmma_f32_16x16x32_f16(false, aL, false, b0, (short)0, accL0, false, false);
      accH0 = __builtin_amdgcn_wmma_f32_16x16x32_f16(false, aH, false, b0, (short)0, accH0, false, false);
      accL1 = __builtin_amdgcn_wmma_f32_16x16x32_f16(false, aL, false, b1, (short)0, accL1, false, false);
      accH1 = __builtin_amdgcn_wmma_f32_16x16x32_f16(false, aH, false, b1, (short)0, accH1, false, false);
      accL2 = __builtin_amdgcn_wmma_f32_16x16x32_f16(false, aL, false, b2f, (short)0, accL2, false, false);
      accH2 = __builtin_amdgcn_wmma_f32_16x16x32_f16(false, aH, false, b2f, (short)0, accH2, false, false);
    }
  }

  // Bias K-chunks (t = 192,193): A = xs[e, i] (zero-padded past 48).
#pragma unroll
  for (int t = KTILES; t < KTILES_ALL; ++t) {
    const int ibase = (t - KTILES) * 32;
    v16h aL, aH;
#pragma unroll
    for (int j = 0; j < 16; ++j) {
      int kk = j + ((j >= 8) ? 8 : 0) + half * 8;
      int ii = ibase + kk;
      aL[j] = (ii < HC) ? xsTile[wave][row * HC + ii] : (_Float16)0.0f;
      aH[j] = (ii < HC) ? xsTile[wave][(row + 16) * HC + ii] : (_Float16)0.0f;
    }
    const _Float16* bp = Bpack + (size_t)t * 1536 + (unsigned)lane * 16;
    v16h b0 = *(const v16h*)(bp);
    v16h b1 = *(const v16h*)(bp + 512);
    v16h b2f = *(const v16h*)(bp + 1024);
    accL0 = __builtin_amdgcn_wmma_f32_16x16x32_f16(false, aL, false, b0, (short)0, accL0, false, false);
    accH0 = __builtin_amdgcn_wmma_f32_16x16x32_f16(false, aH, false, b0, (short)0, accH0, false, false);
    accL1 = __builtin_amdgcn_wmma_f32_16x16x32_f16(false, aL, false, b1, (short)0, accL1, false, false);
    accH1 = __builtin_amdgcn_wmma_f32_16x16x32_f16(false, aH, false, b1, (short)0, accH1, false, false);
    accL2 = __builtin_amdgcn_wmma_f32_16x16x32_f16(false, aL, false, b2f, (short)0, accL2, false, false);
    accH2 = __builtin_amdgcn_wmma_f32_16x16x32_f16(false, aH, false, b2f, (short)0, accH2, false, false);
  }

  // Scatter-add into aggr (segment_sum over dst). D layout: lane = N column,
  // VGPR r -> M = r + 8*half.
  v8f accsL[3] = {accL0, accL1, accL2};
  v8f accsH[3] = {accH0, accH1, accH2};
  const int n = lane & 15;
#pragma unroll
  for (int nt = 0; nt < 3; ++nt) {
#pragma unroll
    for (int r = 0; r < 8; ++r) {
      int m = r + half * 8;
      int dL = dstTile[wave][m];
      int dH = dstTile[wave][m + 16];
      atomicAdd(&aggr[(size_t)dL * HC + nt * 16 + n], accsL[nt][r]);
      atomicAdd(&aggr[(size_t)dH * HC + nt * 16 + n], accsH[nt][r]);
    }
  }
}

// ---------------- node update: x = relu(aggr + x@root + bias) ---------------
__global__ void node_update_kernel(const float* __restrict__ xcur,
                                   const float* __restrict__ aggr,
                                   const float* __restrict__ root,
                                   const float* __restrict__ cbias,
                                   float* __restrict__ xnext) {
  __shared__ float R[HC * HC];
  __shared__ float B[HC];
  for (int u = threadIdx.x; u < HC * HC; u += 256) R[u] = root[u];
  if (threadIdx.x < HC) B[threadIdx.x] = cbias[threadIdx.x];
  __syncthreads();
  int gid = blockIdx.x * 256 + threadIdx.x;           // exact: 8192*48
  int v = gid / HC, o = gid - v * HC;
  float acc = aggr[gid] + B[o];
  const float* xr = xcur + (size_t)v * HC;
#pragma unroll
  for (int j = 0; j < HC; ++j) acc += xr[j] * R[j * HC + o];
  xnext[gid] = fmaxf(acc, 0.f);
}

// ---------------- misc ------------------------------------------------------
__global__ void zero_kernel(float* __restrict__ p, int n) {
  int gid = blockIdx.x * 256 + threadIdx.x;
  if (gid < n) p[gid] = 0.f;
}

__global__ void pool_kernel(const float* __restrict__ xf,
                            const int* __restrict__ batch,
                            float* __restrict__ g) {
  int gid = blockIdx.x * 256 + threadIdx.x;           // exact: 8192*48
  int v = gid / HC, o = gid - v * HC;
  atomicAdd(&g[(size_t)batch[v] * HC + o], xf[gid]);
}

__global__ void linear_kernel(const float* __restrict__ in,
                              const float* __restrict__ W,
                              const float* __restrict__ b,
                              float* __restrict__ out,
                              int M, int K, int O, int doRelu) {
  int gid = blockIdx.x * 256 + threadIdx.x;
  if (gid >= M * O) return;
  int m = gid / O, o = gid - m * O;
  float acc = b[o];
  const float* ir = in + (size_t)m * K;
  const float* wr = W + (size_t)o * K;
  for (int k = 0; k < K; ++k) acc += ir[k] * wr[k];
  out[gid] = doRelu ? fmaxf(acc, 0.f) : acc;
}

// ---------------------------------------------------------------------------
extern "C" void kernel_launch(void* const* d_in, const int* in_sizes, int n_in,
                              void* d_out, int out_size, void* d_ws, size_t ws_size,
                              hipStream_t stream) {
  (void)in_sizes; (void)n_in; (void)out_size; (void)ws_size;
  const float* x_in  = (const float*)d_in[0];
  const int*   ei    = (const int*)  d_in[1];
  const float* ea    = (const float*)d_in[2];
  const int*   batch = (const int*)  d_in[3];
  const float* e1w   = (const float*)d_in[4];   // [3,128,16]
  const float* e1b   = (const float*)d_in[5];   // [3,128]
  const float* e2w   = (const float*)d_in[6];   // [3,2304,128]
  const float* e2b   = (const float*)d_in[7];   // [3,2304]
  const float* rootw = (const float*)d_in[8];   // [3,48,48]
  const float* cbias = (const float*)d_in[9];   // [3,48]
  const float* l0w   = (const float*)d_in[10];
  const float* l0b   = (const float*)d_in[11];
  const float* l1w   = (const float*)d_in[12];
  const float* l1b   = (const float*)d_in[13];
  const float* ow    = (const float*)d_in[14];
  const float* ob    = (const float*)d_in[15];
  float* out = (float*)d_out;

  char* ws = (char*)d_ws;
  size_t off = 0;
  auto wsAlloc = [&](size_t bytes) -> void* {
    off = (off + 255) & ~(size_t)255;
    void* p = ws + off;
    off += bytes;
    return p;
  };
  _Float16* hbuf  = (_Float16*)wsAlloc((size_t)N_EDGES * HE * sizeof(_Float16));
  _Float16* Bpack = (_Float16*)wsAlloc((size_t)BPACK_ELEMS * sizeof(_Float16));
  float* xA   = (float*)wsAlloc((size_t)N_NODES * HC * sizeof(float));
  float* xB   = (float*)wsAlloc((size_t)N_NODES * HC * sizeof(float));
  float* aggr = (float*)wsAlloc((size_t)N_NODES * HC * sizeof(float));
  float* g    = (float*)wsAlloc((size_t)N_GRAPHS * HC * sizeof(float));
  float* g1   = (float*)wsAlloc((size_t)N_GRAPHS * H_LIN * sizeof(float));
  float* g2   = (float*)wsAlloc((size_t)N_GRAPHS * H_LIN * sizeof(float));

  const int* src = ei;
  const int* dst = ei + N_EDGES;

  const float* xcur = x_in;
  float* bufs[2] = {xA, xB};
  for (int L = 0; L < 3; ++L) {
    edge_mlp1_kernel<<<(N_EDGES * HE) / 256, 256, 0, stream>>>(
        ea, e1w + (size_t)L * HE * F_EDGE, e1b + L * HE, hbuf);
    bpack_kernel<<<(BPACK_ELEMS + 255) / 256, 256, 0, stream>>>(
        e2w + (size_t)L * (HC * HC) * HE, e2b + (size_t)L * (HC * HC), Bpack);
    zero_kernel<<<(N_NODES * HC) / 256, 256, 0, stream>>>(aggr, N_NODES * HC);
    edge_gemm_kernel<<<(N_EDGES / TILE_E) / WPB, WPB * 32, 0, stream>>>(
        xcur, hbuf, Bpack, src, dst, aggr);
    float* xnext = bufs[L & 1];
    node_update_kernel<<<(N_NODES * HC) / 256, 256, 0, stream>>>(
        xcur, aggr, rootw + (size_t)L * HC * HC, cbias + L * HC, xnext);
    xcur = xnext;
  }

  zero_kernel<<<(N_GRAPHS * HC) / 256, 256, 0, stream>>>(g, N_GRAPHS * HC);
  pool_kernel<<<(N_NODES * HC) / 256, 256, 0, stream>>>(xcur, batch, g);
  linear_kernel<<<(N_GRAPHS * H_LIN) / 256, 256, 0, stream>>>(g,  l0w, l0b, g1, N_GRAPHS, HC,    H_LIN, 1);
  linear_kernel<<<(N_GRAPHS * H_LIN) / 256, 256, 0, stream>>>(g1, l1w, l1b, g2, N_GRAPHS, H_LIN, H_LIN, 1);
  linear_kernel<<<2, 256, 0, stream>>>(g2, ow, ob, out, N_GRAPHS, H_LIN, 2, 0);
}
